// TinyRelSelfAttention_40879498729278
// MI455X (gfx1250) — compile-verified, matrix operand order checked by reading, not run on hardware
//
#include <hip/hip_runtime.h>
#include <math.h>

// ---------------------------------------------------------------------------
// TinyRelSelfAttention on MI455X (gfx1250): full WMMA pipeline.
// B=2, T=2048, D=1024, H=16, head_dim=64, R=64.
// ---------------------------------------------------------------------------

#define Bn  2
#define Tn  2048
#define Dn  1024
#define Hn  16
#define HDn 64
#define Mn  (Bn * Tn)          // 4096 rows in all projection GEMMs
#define SCALE 0.125f           // 1/sqrt(64)

typedef __attribute__((ext_vector_type(8)))  _Float16 v8h;
typedef __attribute__((ext_vector_type(16))) _Float16 v16h;
typedef __attribute__((ext_vector_type(8)))  float    v8f;

// D = A(16x32 f16) x B(32x16 f16) + C(16x16 f32)
__device__ inline v8f wmma16(v16h a, v16h b, v8f c) {
  return __builtin_amdgcn_wmma_f32_16x16x32_f16(
      /*neg_a=*/false, a, /*neg_b=*/false, b,
      /*c_mod=*/(short)0, c, /*reuse_a=*/false, /*reuse_b=*/false);
}

// Load a 16x32 A-fragment (or mirrored 32x16 B-fragment) for this lane.
// `p` = start of this lane's 32-element contiguous K-slice (row-major memory).
// ISA layout: half h<8 -> K = h + 8*hi ; h>=8 -> K = 16 + (h-8) + 8*hi.
__device__ inline v16h load_frag(const _Float16* p, int hi) {
  union { v16h v; v8h h[2]; } u;
  u.h[0] = *(const v8h*)(p + 8 * hi);
  u.h[1] = *(const v8h*)(p + 16 + 8 * hi);
  return u.v;
}

// max / sum reduction across the 16 lanes that share a C-tile row
__device__ inline float red16_max(float v) {
#pragma unroll
  for (int off = 1; off <= 8; off <<= 1) v = fmaxf(v, __shfl_xor(v, off, 32));
  return v;
}
__device__ inline float red16_sum(float v) {
#pragma unroll
  for (int off = 1; off <= 8; off <<= 1) v += __shfl_xor(v, off, 32);
  return v;
}

// rel-position bias (matches reference: mutually exclusive cases)
__device__ inline float rel_bias(int ri, int rj, float leaf_i, float cross_i,
                                 float cross_j) {
  if (ri == rj) return (ri > 0) ? leaf_i : 0.0f;
  if (ri > 0 && rj == 0) return cross_i;
  if (ri == 0 && rj > 0) return cross_j;
  return 0.0f;
}

// ---------------------------------------------------------------------------
// elementwise f32 -> f16
__global__ __launch_bounds__(256) void cvt_f16_kernel(
    const float* __restrict__ in, _Float16* __restrict__ out, int n) {
  int i = blockIdx.x * 256 + threadIdx.x;
  if (i < n) out[i] = (_Float16)in[i];
}

// W [K=1024, N=1024] f32  ->  Wt [N, K] f16
__global__ __launch_bounds__(256) void transpose_f16_kernel(
    const float* __restrict__ in, _Float16* __restrict__ out) {
  int idx = blockIdx.x * 256 + threadIdx.x;   // n*1024 + k, k fastest
  int k = idx & (Dn - 1);
  int nn = idx >> 10;
  out[idx] = (_Float16)in[(size_t)k * Dn + nn];
}

// ---------------------------------------------------------------------------
// Generic WMMA GEMM:  C[M=4096, N=1024] = A[M,K=1024] x Wt[N,K]^T + bias[N]
// Block = 8 waves; wave tile 16(M) x 64(N); block tile 128 x 64.
// MODE 0: f32 out, row-major [M,N]            (final projection)
// MODE 1: f16 out, scattered to [B*H, T, 64]  (Q, K)
// MODE 2: f16 out, scattered to [B*H, 64, T]  (V transposed)
template <int MODE>
__global__ __launch_bounds__(256) void gemm_wmma_kernel(
    const _Float16* __restrict__ A, const _Float16* __restrict__ Wt,
    const float* __restrict__ bias, void* __restrict__ outp) {
  const int wave = threadIdx.x >> 5;
  const int lane = threadIdx.x & 31;
  const int hi = lane >> 4;
  const int n = lane & 15;

  const int mb = blockIdx.x & 31;           // 4096/128 = 32 M-tiles
  const int nb = blockIdx.x >> 5;           // 1024/64  = 16 N-tiles
  const int mBase = mb * 128 + wave * 16;
  const int nBase = nb * 64;

  const _Float16* aRow = A + (size_t)(mBase + n) * Dn;   // A-frag row
  const _Float16* w0 = Wt + (size_t)(nBase + 0 + n) * Dn;
  const _Float16* w1 = Wt + (size_t)(nBase + 16 + n) * Dn;
  const _Float16* w2 = Wt + (size_t)(nBase + 32 + n) * Dn;
  const _Float16* w3 = Wt + (size_t)(nBase + 48 + n) * Dn;

  v8f acc0 = {}, acc1 = {}, acc2 = {}, acc3 = {};
#pragma unroll 4
  for (int kk = 0; kk < Dn; kk += 32) {
    v16h a = load_frag(aRow + kk, hi);
    v16h b0 = load_frag(w0 + kk, hi);
    v16h b1 = load_frag(w1 + kk, hi);
    v16h b2 = load_frag(w2 + kk, hi);
    v16h b3 = load_frag(w3 + kk, hi);
    acc0 = wmma16(a, b0, acc0);
    acc1 = wmma16(a, b1, acc1);
    acc2 = wmma16(a, b2, acc2);
    acc3 = wmma16(a, b3, acc3);
  }

  v8f accs[4] = {acc0, acc1, acc2, acc3};
#pragma unroll
  for (int s = 0; s < 4; ++s) {
    const int col = nBase + s * 16 + n;
    const float bval = bias[col];
#pragma unroll
    for (int e = 0; e < 8; ++e) {
      const int row = mBase + e + 8 * hi;     // C layout: M = e + 8*hi
      const float val = accs[s][e] + bval;
      if (MODE == 0) {
        ((float*)outp)[(size_t)row * Dn + col] = val;
      } else {
        const int b = row >> 11;              // row / T
        const int t = row & (Tn - 1);
        const int h = col >> 6;               // col / 64
        const int hd = col & 63;
        _Float16* o = (_Float16*)outp;
        if (MODE == 1)
          o[(((size_t)(b * Hn + h)) * Tn + t) * HDn + hd] = (_Float16)val;
        else
          o[(((size_t)(b * Hn + h)) * HDn + hd) * Tn + t] = (_Float16)val;
      }
    }
  }
}

// ---------------------------------------------------------------------------
// Flash attention: block = 8 waves = 128 queries of one (b,h); keys in
// blocks of 32. grid = B*H * (T/128) = 512.
__global__ __launch_bounds__(256) void attn_wmma_kernel(
    const _Float16* __restrict__ q16, const _Float16* __restrict__ k16,
    const _Float16* __restrict__ vT, const int* __restrict__ rel_ids,
    const int* __restrict__ attn_mask, const float* __restrict__ bias_leaf,
    const float* __restrict__ bias_cross, _Float16* __restrict__ o16) {
  __shared__ _Float16 pbuf[8][16][40];   // per-wave P tile, padded rows

  const int wave = threadIdx.x >> 5;
  const int lane = threadIdx.x & 31;
  const int hi = lane >> 4;
  const int n = lane & 15;

  const int bh = blockIdx.x >> 4;        // 16 query-blocks per (b,h)
  const int qb = blockIdx.x & 15;
  const int b = bh >> 4;                 // H = 16
  const int h = bh & 15;
  const int t0 = qb * 128 + wave * 16;

  // Q A-fragments (persist across key loop)
  const _Float16* qRow = q16 + ((size_t)bh * Tn + t0 + n) * HDn;
  const v16h aQ0 = load_frag(qRow, hi);        // hd 0..31
  const v16h aQ1 = load_frag(qRow + 32, hi);   // hd 32..63

  // per-row rel-id state (row m = e + 8*hi of the C tile)
  int ri[8];
  float leaf_i[8], cross_i[8], mrun[8], lrun[8];
#pragma unroll
  for (int e = 0; e < 8; ++e) {
    const int t = t0 + e + 8 * hi;
    ri[e] = rel_ids[b * Tn + t];
    leaf_i[e] = bias_leaf[ri[e]];
    cross_i[e] = bias_cross[ri[e]];
    mrun[e] = -1e30f;
    lrun[e] = 0.0f;
  }
  v8f acc0 = {}, acc1 = {}, acc2 = {}, acc3 = {};

  const _Float16* vRow0 = vT + ((size_t)bh * HDn + 0 + n) * Tn;
  const _Float16* vRow1 = vT + ((size_t)bh * HDn + 16 + n) * Tn;
  const _Float16* vRow2 = vT + ((size_t)bh * HDn + 32 + n) * Tn;
  const _Float16* vRow3 = vT + ((size_t)bh * HDn + 48 + n) * Tn;

  for (int j0 = 0; j0 < Tn; j0 += 32) {
    // ---- S = Q K^T (two 16x16 tiles, K-dim = head_dim 64) ----
    const _Float16* kRow0 = k16 + ((size_t)bh * Tn + j0 + n) * HDn;
    const _Float16* kRow1 = kRow0 + 16 * HDn;
    v16h bK0a = load_frag(kRow0, hi), bK0b = load_frag(kRow0 + 32, hi);
    v16h bK1a = load_frag(kRow1, hi), bK1b = load_frag(kRow1 + 32, hi);
    v8f z = {};
    v8f s0 = wmma16(aQ0, bK0a, wmma16(aQ1, bK0b, z));
    v8f s1 = wmma16(aQ0, bK1a, wmma16(aQ1, bK1b, z));

    // ---- bias + mask + online softmax ----
    const int jA = j0 + n, jB = j0 + 16 + n;    // this lane's key columns
    const int rjA = rel_ids[b * Tn + jA];
    const int rjB = rel_ids[b * Tn + jB];
    const float cjA = bias_cross[rjA], cjB = bias_cross[rjB];
    const bool mskA = attn_mask[b * Tn + jA] != 0;
    const bool mskB = attn_mask[b * Tn + jB] != 0;

    float p0[8], p1[8], alpha[8];
#pragma unroll
    for (int e = 0; e < 8; ++e) {
      float v0 = s0[e] * SCALE + rel_bias(ri[e], rjA, leaf_i[e], cross_i[e], cjA);
      float v1 = s1[e] * SCALE + rel_bias(ri[e], rjB, leaf_i[e], cross_i[e], cjB);
      v0 = mskA ? v0 : -1e30f;
      v1 = mskB ? v1 : -1e30f;
      const float mx = red16_max(fmaxf(v0, v1));
      const float mnew = fmaxf(mrun[e], mx);
      alpha[e] = expf(mrun[e] - mnew);
      p0[e] = expf(v0 - mnew);
      p1[e] = expf(v1 - mnew);
      lrun[e] = lrun[e] * alpha[e] + red16_sum(p0[e] + p1[e]);
      mrun[e] = mnew;
    }
#pragma unroll
    for (int e = 0; e < 8; ++e) {
      acc0[e] *= alpha[e]; acc1[e] *= alpha[e];
      acc2[e] *= alpha[e]; acc3[e] *= alpha[e];
    }

    // ---- P (C-layout) -> LDS -> A-fragment layout ----
    __syncthreads();
#pragma unroll
    for (int e = 0; e < 8; ++e) {
      pbuf[wave][e + 8 * hi][n] = (_Float16)p0[e];
      pbuf[wave][e + 8 * hi][16 + n] = (_Float16)p1[e];
    }
    __syncthreads();
    const v16h aP = load_frag(&pbuf[wave][n][0], hi);

    // ---- O += P x V (K-dim = 32 local keys) ----
    acc0 = wmma16(aP, load_frag(vRow0 + j0, hi), acc0);
    acc1 = wmma16(aP, load_frag(vRow1 + j0, hi), acc1);
    acc2 = wmma16(aP, load_frag(vRow2 + j0, hi), acc2);
    acc3 = wmma16(aP, load_frag(vRow3 + j0, hi), acc3);
  }

  // ---- normalize + store O as f16 rows of [B*T, D] for the final GEMM ----
  v8f accs[4] = {acc0, acc1, acc2, acc3};
#pragma unroll
  for (int s = 0; s < 4; ++s) {
#pragma unroll
    for (int e = 0; e < 8; ++e) {
      const int t = t0 + e + 8 * hi;
      const int col = h * HDn + s * 16 + n;
      const float val = accs[s][e] / lrun[e];
      o16[((size_t)(b * Tn + t)) * Dn + col] = (_Float16)val;
    }
  }
}

// ---------------------------------------------------------------------------
extern "C" void kernel_launch(void* const* d_in, const int* in_sizes, int n_in,
                              void* d_out, int out_size, void* d_ws,
                              size_t ws_size, hipStream_t stream) {
  const float* x          = (const float*)d_in[0];
  const int*   attn_mask  = (const int*)d_in[1];
  const int*   rel_ids    = (const int*)d_in[2];
  const float* Wq         = (const float*)d_in[3];
  const float* bq         = (const float*)d_in[4];
  const float* Wk         = (const float*)d_in[5];
  const float* bk         = (const float*)d_in[6];
  const float* Wv         = (const float*)d_in[7];
  const float* bv         = (const float*)d_in[8];
  const float* Wo         = (const float*)d_in[9];
  const float* bo         = (const float*)d_in[10];
  const float* bias_leaf  = (const float*)d_in[11];
  const float* bias_cross = (const float*)d_in[12];

  // workspace layout (48 MiB total)
  char* ws = (char*)d_ws;
  _Float16* x16  = (_Float16*)(ws + (size_t)0);
  _Float16* WqT  = (_Float16*)(ws + ((size_t)8  << 20));
  _Float16* WkT  = (_Float16*)(ws + ((size_t)10 << 20));
  _Float16* WvT  = (_Float16*)(ws + ((size_t)12 << 20));
  _Float16* WoT  = (_Float16*)(ws + ((size_t)14 << 20));
  _Float16* q16  = (_Float16*)(ws + ((size_t)16 << 20));
  _Float16* k16  = (_Float16*)(ws + ((size_t)24 << 20));
  _Float16* v16T = (_Float16*)(ws + ((size_t)32 << 20));
  _Float16* o16  = (_Float16*)(ws + ((size_t)40 << 20));

  const int nx = Mn * Dn;                         // 4,194,304
  cvt_f16_kernel<<<nx / 256, 256, 0, stream>>>(x, x16, nx);
  transpose_f16_kernel<<<(Dn * Dn) / 256, 256, 0, stream>>>(Wq, WqT);
  transpose_f16_kernel<<<(Dn * Dn) / 256, 256, 0, stream>>>(Wk, WkT);
  transpose_f16_kernel<<<(Dn * Dn) / 256, 256, 0, stream>>>(Wv, WvT);
  transpose_f16_kernel<<<(Dn * Dn) / 256, 256, 0, stream>>>(Wo, WoT);

  const int gemmBlocks = (Mn / 128) * (Dn / 64);  // 512
  gemm_wmma_kernel<1><<<gemmBlocks, 256, 0, stream>>>(x16, WqT, bq, q16);
  gemm_wmma_kernel<1><<<gemmBlocks, 256, 0, stream>>>(x16, WkT, bk, k16);
  gemm_wmma_kernel<2><<<gemmBlocks, 256, 0, stream>>>(x16, WvT, bv, v16T);

  attn_wmma_kernel<<<Bn * Hn * (Tn / 128), 256, 0, stream>>>(
      q16, k16, v16T, rel_ids, attn_mask, bias_leaf, bias_cross, o16);

  gemm_wmma_kernel<0><<<gemmBlocks, 256, 0, stream>>>(o16, WoT, bo, d_out);
}
